// AttentionWithFastKAN_72627896975544
// MI455X (gfx1250) — matrix operand order repaired and checked by reading, not compiled, not run on hardware
//
#include <hip/hip_runtime.h>
#include <hip/hip_bf16.h>

// ---------------------------------------------------------------------------
// Problem constants
// ---------------------------------------------------------------------------
#define BB   8
#define NN   1024
#define CC   768
#define HH   12
#define DD   64
#define GG   8
#define MM   (BB * NN)          // 8192 token rows
#define KTOT (CC * GG + CC)     // 6912 = basis(6144) + silu(768)
#define NQKV (3 * CC)           // 2304
#define NPRJ CC                 // 768

typedef __attribute__((ext_vector_type(16))) __bf16 v16bf;
typedef __attribute__((ext_vector_type(8)))  float  v8f;

union FragBF { uint4 u[2]; v16bf v; };

__device__ __forceinline__ unsigned short f2bf(float f) {
    unsigned int u = __float_as_uint(f);
    u += 0x7FFFu + ((u >> 16) & 1u);      // round-to-nearest-even
    return (unsigned short)(u >> 16);
}
__device__ __forceinline__ float bf2f(unsigned short h) {
    return __uint_as_float(((unsigned int)h) << 16);
}

__device__ __forceinline__ v8f wmma_bf16(const FragBF& a, const FragBF& b, v8f c) {
    return __builtin_amdgcn_wmma_f32_16x16x32_bf16(
        false, a.v, false, b.v, (short)0, c, false, false);
}

// ---------------------------------------------------------------------------
// Weight prep: Wt[n][k] = bf16( k<6144 ? spline[k][n] : base[k-6144][n] )
// (transposed so GEMM B-fragments read contiguous K per output column)
// ---------------------------------------------------------------------------
__global__ void kan_prep_w(const float* __restrict__ spline,
                           const float* __restrict__ base,
                           unsigned short* __restrict__ Wt, int Nout) {
    size_t i = (size_t)blockIdx.x * 256 + threadIdx.x;
    size_t total = (size_t)Nout * KTOT;
    if (i >= total) return;
    int n = (int)(i % Nout);
    int k = (int)(i / Nout);
    float v = (k < CC * GG) ? spline[(size_t)k * Nout + n]
                            : base[(size_t)(k - CC * GG) * Nout + n];
    Wt[(size_t)n * KTOT + k] = f2bf(v);
}

// ---------------------------------------------------------------------------
// Activations: layernorm -> RBF basis (8 grids) + silu, bf16 out [M, 6912]
// One block (256 thr) per token row.
// ---------------------------------------------------------------------------
__global__ __launch_bounds__(256) void kan_act(const float* __restrict__ X,
                                               const float* __restrict__ lnw,
                                               const float* __restrict__ lnb,
                                               unsigned short* __restrict__ Act) {
    __shared__ float r1[256], r2[256];
    int m = blockIdx.x;
    int tid = threadIdx.x;
    const float* x = X + (size_t)m * CC;

    float xv[3];
    float s = 0.f, ss = 0.f;
#pragma unroll
    for (int i = 0; i < 3; i++) {
        float v = x[tid + 256 * i];
        xv[i] = v; s += v; ss += v * v;
    }
    r1[tid] = s; r2[tid] = ss;
    __syncthreads();
    for (int o = 128; o > 0; o >>= 1) {
        if (tid < o) { r1[tid] += r1[tid + o]; r2[tid] += r2[tid + o]; }
        __syncthreads();
    }
    float mu  = r1[0] * (1.f / CC);
    float var = r2[0] * (1.f / CC) - mu * mu;
    float rs  = rsqrtf(var + 1e-5f);

    unsigned short* act = Act + (size_t)m * KTOT;
#pragma unroll
    for (int i = 0; i < 3; i++) {
        int c = tid + 256 * i;
        float h = (xv[i] - mu) * rs * lnw[c] + lnb[c];
#pragma unroll
        for (int g = 0; g < GG; g++) {
            float gp = -2.f + g * (4.f / 7.f);
            float t  = (h - gp) * (7.f / 4.f);   // 1/DENOM = 7/4
            act[c * GG + g] = f2bf(__expf(-t * t));
        }
        float sil = xv[i] / (1.f + __expf(-xv[i]));
        act[CC * GG + c] = f2bf(sil);
    }
}

// ---------------------------------------------------------------------------
// GEMM: Out[M,Nout] = Act[M,K] (bf16) @ Wt[Nout,K]^T (bf16) + bias, f32 out
// 256 thr = 8 waves (2x4), block tile 128x128, wave tile 64x32, BK=32.
// LDS double-buffered; tiles filled with GLOBAL_LOAD_ASYNC_TO_LDS_B128
// (ASYNCcnt-tracked, no VGPR staging), one barrier per k-step.
// ---------------------------------------------------------------------------
#define GBM 128
#define GBN 128
#define GBK 32
#define LPAD 40   // row stride (elements); 80B keeps 16B alignment

__global__ __launch_bounds__(256) void kan_gemm(const unsigned short* __restrict__ A,
                                                const unsigned short* __restrict__ Wt,
                                                const float* __restrict__ bias,
                                                float* __restrict__ Out, int Nout) {
    __shared__ unsigned short As[2][GBM][LPAD];   // 2 x 10240 B
    __shared__ unsigned short Bs[2][GBN][LPAD];   // 2 x 10240 B

    int tid  = threadIdx.x;
    int bm   = blockIdx.y * GBM;
    int bn   = blockIdx.x * GBN;
    int w    = tid >> 5, lane = tid & 31;
    int wr   = w & 1, wc = w >> 1;       // 2 x 4 wave grid, wave tile 64x32
    int half = lane >> 4, l16 = lane & 15;

    v8f acc[4][2] = {};

    // Each thread moves one 32B chunk of A and of B per k-step.
    int row  = tid >> 1, colh = (tid & 1) * 16;
    const unsigned short* Ag = A  + (size_t)(bm + row) * KTOT + colh;
    const unsigned short* Bg = Wt + (size_t)(bn + row) * KTOT + colh;
    // Low 32 bits of a generic LDS pointer == byte offset within the LDS
    // allocation (aperture mapping truncates upper bits).
    unsigned ldsA0 = (unsigned)(unsigned long long)&As[0][row][colh];
    unsigned ldsB0 = (unsigned)(unsigned long long)&Bs[0][row][colh];
    const unsigned bufStride = (unsigned)sizeof(As[0]);   // 10240 B

    auto issue_async = [&](int kc, int nb) {
        unsigned la = ldsA0 + (unsigned)nb * bufStride;
        unsigned lb = ldsB0 + (unsigned)nb * bufStride;
        unsigned long long ga = (unsigned long long)(Ag + kc);
        unsigned long long gb = (unsigned long long)(Bg + kc);
        // INST_OFFSET applies to both LDS and global addresses.
        asm volatile(
            "global_load_async_to_lds_b128 %0, %2, off\n\t"
            "global_load_async_to_lds_b128 %0, %2, off offset:16\n\t"
            "global_load_async_to_lds_b128 %1, %3, off\n\t"
            "global_load_async_to_lds_b128 %1, %3, off offset:16"
            :: "v"(la), "v"(lb), "v"(ga), "v"(gb) : "memory");
    };

    issue_async(0, 0);
    asm volatile("s_wait_asynccnt 0" ::: "memory");
    __syncthreads();

    int buf = 0;
    for (int kc = 0; kc < KTOT; kc += GBK) {
        int nb = buf ^ 1;
        if (kc + GBK < KTOT) issue_async(kc + GBK, nb);   // prefetch next tile

        FragBF a[4], b[2];
#pragma unroll
        for (int mi = 0; mi < 4; mi++) {
            const unsigned short* p = &As[buf][wr * 64 + mi * 16 + l16][half * 8];
            a[mi].u[0] = *(const uint4*)p;
            a[mi].u[1] = *(const uint4*)(p + 16);
        }
#pragma unroll
        for (int ni = 0; ni < 2; ni++) {
            const unsigned short* p = &Bs[buf][wc * 32 + ni * 16 + l16][half * 8];
            b[ni].u[0] = *(const uint4*)p;
            b[ni].u[1] = *(const uint4*)(p + 16);
        }
#pragma unroll
        for (int mi = 0; mi < 4; mi++)
#pragma unroll
            for (int ni = 0; ni < 2; ni++)
                acc[mi][ni] = wmma_bf16(a[mi], b[ni], acc[mi][ni]);

        asm volatile("s_wait_asynccnt 0" ::: "memory");
        __syncthreads();
        buf = nb;
    }

#pragma unroll
    for (int mi = 0; mi < 4; mi++)
#pragma unroll
        for (int ni = 0; ni < 2; ni++) {
            int n = bn + wc * 32 + ni * 16 + l16;
            float bi = bias[n];
#pragma unroll
            for (int j = 0; j < 8; j++) {
                int m = bm + wr * 64 + mi * 16 + j + 8 * half;
                Out[(size_t)m * Nout + n] = acc[mi][ni][j] + bi;
            }
        }
}

// ---------------------------------------------------------------------------
// Repack qkv f32 -> Q,K bf16 [B,H,N,D] and V^T bf16 [B,H,D,N]
// ---------------------------------------------------------------------------
__global__ void kan_repack(const float* __restrict__ qkv,
                           unsigned short* __restrict__ Qb,
                           unsigned short* __restrict__ Kb,
                           unsigned short* __restrict__ Vt) {
    size_t i = (size_t)blockIdx.x * 256 + threadIdx.x;
    if (i >= (size_t)BB * HH * NN * DD) return;
    int d  = (int)(i & 63);
    int n  = (int)((i >> 6) & 1023);
    int bh = (int)(i >> 16);             // b*12 + h
    int h  = bh % HH, b = bh / HH;
    size_t base = ((size_t)(b * NN + n)) * NQKV + h * DD + d;
    Qb[i] = f2bf(qkv[base]);
    Kb[i] = f2bf(qkv[base + CC]);
    Vt[(((size_t)bh) * DD + d) * NN + n] = f2bf(qkv[base + 2 * CC]);
}

// ---------------------------------------------------------------------------
// Attention: one wave per 16 query rows of one (b,h). WMMA for QK^T and PV.
// Scores kept as bf16 in 32KB LDS; softmax pair-reduced with shfl_xor(16).
// grid = (64, 12, 8), block = 32.
// ---------------------------------------------------------------------------
__global__ __launch_bounds__(32) void kan_attn(const unsigned short* __restrict__ Qb,
                                               const unsigned short* __restrict__ Kb,
                                               const unsigned short* __restrict__ Vt,
                                               float* __restrict__ Out) {
    __shared__ unsigned short S[16 * NN];   // 32 KB bf16 score/prob rows
    __shared__ float invbuf[16];

    int lane = threadIdx.x;
    int half = lane >> 4, l16 = lane & 15;
    int bh = blockIdx.z * HH + blockIdx.y;  // b*12 + h
    int m0 = blockIdx.x * 16;               // query row block within sequence
    const unsigned short* Qbase = Qb + ((size_t)bh * NN) * DD;
    const unsigned short* Kbase = Kb + ((size_t)bh * NN) * DD;
    const unsigned short* Vbase = Vt + ((size_t)bh * DD) * NN;
    const float scale = 0.125f;             // 64^-0.5

    // --- Phase 1: S = scale * Q @ K^T ------------------------------------
    FragBF afr[2];
    {
        const unsigned short* qrow = Qbase + (size_t)(m0 + l16) * DD;
#pragma unroll
        for (int kc = 0; kc < 2; kc++) {
            const unsigned short* pa = qrow + kc * 32 + half * 8;
            afr[kc].u[0] = *(const uint4*)pa;
            afr[kc].u[1] = *(const uint4*)(pa + 16);
        }
    }
    for (int nt = 0; nt < NN / 16; nt++) {
        v8f acc = {};
        const unsigned short* krow = Kbase + (size_t)(nt * 16 + l16) * DD;
#pragma unroll
        for (int kc = 0; kc < 2; kc++) {
            FragBF b;
            const unsigned short* pb = krow + kc * 32 + half * 8;
            b.u[0] = *(const uint4*)pb;
            b.u[1] = *(const uint4*)(pb + 16);
            acc = wmma_bf16(afr[kc], b, acc);
        }
#pragma unroll
        for (int j = 0; j < 8; j++) {
            int ml = j + 8 * half;
            S[ml * NN + nt * 16 + l16] = f2bf(acc[j] * scale);
        }
    }
    __syncthreads();

    // --- Phase 2: softmax (row r split across lane pair r / r+16) --------
    {
        unsigned short* srow = S + l16 * NN + half * 512;
        float mx = -1e30f;
        for (int i = 0; i < 512; i++) mx = fmaxf(mx, bf2f(srow[i]));
        mx = fmaxf(mx, __shfl_xor(mx, 16, 32));
        float sum = 0.f;
        for (int i = 0; i < 512; i++) {
            float e = __expf(bf2f(srow[i]) - mx);
            sum += e;
            srow[i] = f2bf(e);              // in-place, same slot: no overlap
        }
        sum += __shfl_xor(sum, 16, 32);
        if (half == 0) invbuf[l16] = 1.f / sum;
    }
    __syncthreads();

    // --- Phase 3: O = (P @ V) * inv_sum ----------------------------------
    int h = bh % HH, b = bh / HH;
#pragma unroll
    for (int nt = 0; nt < DD / 16; nt++) {
        v8f acc = {};
        const unsigned short* vrow = Vbase + (size_t)(nt * 16 + l16) * NN;
        const unsigned short* prow = S + l16 * NN;
        for (int kc = 0; kc < NN / 32; kc++) {
            FragBF a, vb;
            const unsigned short* pa = prow + kc * 32 + half * 8;
            a.u[0] = *(const uint4*)pa;
            a.u[1] = *(const uint4*)(pa + 16);
            const unsigned short* pb = vrow + kc * 32 + half * 8;
            vb.u[0] = *(const uint4*)pb;
            vb.u[1] = *(const uint4*)(pb + 16);
            acc = wmma_bf16(a, vb, acc);
        }
#pragma unroll
        for (int j = 0; j < 8; j++) {
            int ml = j + 8 * half;
            int row = b * NN + m0 + ml;
            int col = h * DD + nt * 16 + l16;
            Out[(size_t)row * CC + col] = acc[j] * invbuf[ml];
        }
    }
}

// ---------------------------------------------------------------------------
// Host launcher
// ---------------------------------------------------------------------------
extern "C" void kernel_launch(void* const* d_in, const int* in_sizes, int n_in,
                              void* d_out, int out_size, void* d_ws, size_t ws_size,
                              hipStream_t stream) {
    const float* x        = (const float*)d_in[0];
    const float* q_lnw    = (const float*)d_in[1];
    const float* q_lnb    = (const float*)d_in[2];
    const float* q_spline = (const float*)d_in[3];
    const float* q_basew  = (const float*)d_in[4];
    const float* q_baseb  = (const float*)d_in[5];
    const float* p_lnw    = (const float*)d_in[6];
    const float* p_lnb    = (const float*)d_in[7];
    const float* p_spline = (const float*)d_in[8];
    const float* p_basew  = (const float*)d_in[9];
    const float* p_baseb  = (const float*)d_in[10];

    char* ws = (char*)d_ws;
    size_t off = 0;
    auto alloc = [&](size_t bytes) -> void* {
        void* p = ws + off;
        off += (bytes + 255) & ~(size_t)255;
        return p;
    };
    unsigned short* Act  = (unsigned short*)alloc((size_t)MM * KTOT * 2);
    unsigned short* WtQ  = (unsigned short*)alloc((size_t)NQKV * KTOT * 2);
    unsigned short* WtP  = (unsigned short*)alloc((size_t)NPRJ * KTOT * 2);
    float*          qkvB = (float*)alloc((size_t)MM * NQKV * 4);
    unsigned short* Qb   = (unsigned short*)alloc((size_t)BB * HH * NN * DD * 2);
    unsigned short* Kb   = (unsigned short*)alloc((size_t)BB * HH * NN * DD * 2);
    unsigned short* Vt   = (unsigned short*)alloc((size_t)BB * HH * NN * DD * 2);
    float*          aOut = (float*)alloc((size_t)MM * CC * 4);

    // 1) weights -> bf16 transposed concat
    {
        size_t tq = (size_t)NQKV * KTOT, tp = (size_t)NPRJ * KTOT;
        kan_prep_w<<<(unsigned)((tq + 255) / 256), 256, 0, stream>>>(q_spline, q_basew, WtQ, NQKV);
        kan_prep_w<<<(unsigned)((tp + 255) / 256), 256, 0, stream>>>(p_spline, p_basew, WtP, NPRJ);
    }
    // 2) qkv FastKAN
    kan_act<<<MM, 256, 0, stream>>>(x, q_lnw, q_lnb, Act);
    kan_gemm<<<dim3(NQKV / GBN, MM / GBM), 256, 0, stream>>>(Act, WtQ, q_baseb, qkvB, NQKV);
    // 3) attention
    {
        size_t te = (size_t)BB * HH * NN * DD;
        kan_repack<<<(unsigned)((te + 255) / 256), 256, 0, stream>>>(qkvB, Qb, Kb, Vt);
    }
    kan_attn<<<dim3(NN / 16, HH, BB), 32, 0, stream>>>(Qb, Kb, Vt, aOut);
    // 4) proj FastKAN -> d_out
    kan_act<<<MM, 256, 0, stream>>>(aOut, p_lnw, p_lnb, Act);
    kan_gemm<<<dim3(NPRJ / GBN, MM / GBM), 256, 0, stream>>>(Act, WtP, p_baseb, (float*)d_out, NPRJ);

    (void)in_sizes; (void)n_in; (void)out_size; (void)ws_size;
}
